// EfficientMoE_56444460204382
// MI455X (gfx1250) — matrix operand order repaired and checked by reading, not compiled, not run on hardware
//
#include <hip/hip_runtime.h>

// ---------------------------------------------------------------------------
// EfficientMoE for MI455X (gfx1250): bf16 WMMA pipeline
//   x:(16,512,4,4096,1) f32 -> out:(16,16,4,4096,1) f32
// ---------------------------------------------------------------------------

typedef __bf16 v16bf __attribute__((ext_vector_type(16)));
typedef float  v8f   __attribute__((ext_vector_type(8)));
typedef float  v4f   __attribute__((ext_vector_type(4)));

#define E_   4
#define M_   4
#define C_   512
#define H1_  128
#define H2_  48
#define H2P_ 64
#define OUT_ 16
#define B_   16
#define N_   4096
#define NTILE_ 128   // n-columns per block

// ---- dynamic LDS layout (bytes) -------------------------------------------
#define XP_OFF   0u
#define XP_SZ    (512u * 128u * 2u)      // X tile, bf16 B-fragment packed (128 KB)
#define H1P_OFF  (XP_OFF + XP_SZ)
#define H1P_SZ   (128u * 128u * 2u)      // h1 tile (32 KB)
#define H2P_OFF  (H1P_OFF + H1P_SZ)
#define H2P_SZ   (64u * 128u * 2u)       // h2 tile, padded to 64 rows (16 KB)
#define SMEM_BYTES (XP_SZ + H1P_SZ + H2P_SZ)   // 176 KB (<320 KB/WGP)

// ---- workspace layout (bytes) ---------------------------------------------
#define WS_PW1    0u
#define WS_PW1_SZ (4u * 8u * 16u * 1024u)   // E x 8 Mtiles x 16 Ktiles x 1KB
#define WS_PW2    (WS_PW1 + WS_PW1_SZ)
#define WS_PW2_SZ (4u * 4u * 4u * 1024u)
#define WS_PW3    (WS_PW2 + WS_PW2_SZ)
#define WS_PW3_SZ (4u * 1u * 2u * 1024u)
#define WS_POOL   (WS_PW3 + WS_PW3_SZ)
#define WS_POOL_SZ (16u * 512u * 4u * 4u)   // pooled (b,c,m) f32
#define WS_G      (WS_POOL + WS_POOL_SZ)
#define WS_G_SZ   (16u * 4u * 4u * 4u)      // gate logits (b,E,M) f32
#define WS_WTS    (WS_G + WS_G_SZ)          // softmax weights (b,4,4) f32

__device__ __forceinline__ unsigned pack_bf16x2(float a, float b) {
  unsigned short ua = __builtin_bit_cast(unsigned short, (__bf16)a);
  unsigned short ub = __builtin_bit_cast(unsigned short, (__bf16)b);
  return (unsigned)ua | ((unsigned)ub << 16);
}

__device__ __forceinline__ v8f wmma_bf16(v16bf a, v16bf b, v8f c) {
  return __builtin_amdgcn_wmma_f32_16x16x32_bf16(false, a, false, b,
                                                 (short)0, c, false, false);
}

// ---------------------------------------------------------------------------
// Repack f32 weights (E, realM, realK) into bf16 WMMA A-fragment tiles.
// Tile = 16x32, 1024B; lane l holds row M = mi*16+(l&15):
//   l<16 : elements 0..15 = K { 0..15 }
//   l>=16: elements 0..15 = K { 16..31 }
// Out-of-range M/K -> 0 (pads H2 48->64 and W3's K 48->64).
// ---------------------------------------------------------------------------
__global__ void pack_w(const float* __restrict__ src, __bf16* __restrict__ dst,
                       int numM, int numK, int realM, int realK) {
  int gid = blockIdx.x * 256 + threadIdx.x;
  int total = E_ * numM * numK * 512;
  if (gid >= total) return;
  int el   = gid & 511;
  int tile = gid >> 9;
  int ki = tile % numK;
  int mi = (tile / numK) % numM;
  int e  = tile / (numK * numM);
  int lane = el >> 4;
  int r    = el & 15;
  int v    = r >> 1;
  int hf   = r & 1;
  int Mrow = mi * 16 + (lane & 15);
  int kl = 2 * v + hf;
  if (lane >= 16) kl += 16;
  int K = ki * 32 + kl;
  float f = 0.f;
  if (Mrow < realM && K < realK)
    f = src[((size_t)e * realM + Mrow) * realK + K];
  dst[gid] = (__bf16)f;
}

// ---------------------------------------------------------------------------
// pooled[b,c,m] = mean_n x[b,c,m,n]; one wave32 per row.
// ---------------------------------------------------------------------------
__global__ void pool_k(const float* __restrict__ x, float* __restrict__ pooled) {
  int w = threadIdx.x >> 5, lane = threadIdx.x & 31;
  int r = blockIdx.x * 8 + w;               // (b*512+c)*4+m, 32768 rows
  size_t base = (size_t)r * 4096;
  float s = 0.f;
  #pragma unroll 4
  for (int i = 0; i < 32; ++i) {
    v4f t = *(const v4f*)(x + base + (size_t)(lane + 32 * i) * 4);
    s += t[0] + t[1] + t[2] + t[3];
  }
  for (int off = 16; off > 0; off >>= 1) s += __shfl_xor(s, off, 32);
  if (lane == 0) pooled[r] = s * (1.f / 4096.f);
}

// ---------------------------------------------------------------------------
// Gate MLP: g[b,e,m] = Wg2 @ relu(Wg1 @ pooled + bg1) + bg2. Block=(b,m).
// ---------------------------------------------------------------------------
__global__ void gate_k(const float* __restrict__ pooled,
                       const float* __restrict__ Wg1, const float* __restrict__ bg1,
                       const float* __restrict__ Wg2, const float* __restrict__ bg2,
                       float* __restrict__ gbuf) {
  __shared__ float g1s[128];
  int bm = blockIdx.x;          // 0..63
  int b = bm >> 2, m = bm & 3;
  int o = threadIdx.x;          // 0..127
  const float* wrow = Wg1 + (size_t)o * 512;
  const float* p = pooled + (size_t)b * 2048 + m;   // stride 4 over c
  float s = 0.f;
  for (int c = 0; c < 512; ++c) s += wrow[c] * p[c * 4];
  g1s[o] = fmaxf(s + bg1[o], 0.f);
  __syncthreads();
  if (o < 4) {
    float t = bg2[o];
    for (int i = 0; i < 128; ++i) t += Wg2[o * 128 + i] * g1s[i];
    gbuf[(b * 4 + o) * 4 + m] = t;   // (b, E, M) row-major
  }
}

// Reference reshape (b,E,M)->(b,M,E) is a raw reshape; with E==M==4 it is the
// identity on indices, so weights[b,m,e] = softmax_e g[b, m(axisE), e(axisM)].
__global__ void softmax_k(const float* __restrict__ gbuf, float* __restrict__ wts) {
  int t = threadIdx.x;          // 64 rows of 4
  float a0 = gbuf[t * 4 + 0], a1 = gbuf[t * 4 + 1];
  float a2 = gbuf[t * 4 + 2], a3 = gbuf[t * 4 + 3];
  float mx = fmaxf(fmaxf(a0, a1), fmaxf(a2, a3));
  float e0 = __expf(a0 - mx), e1 = __expf(a1 - mx);
  float e2 = __expf(a2 - mx), e3 = __expf(a3 - mx);
  float inv = 1.f / (e0 + e1 + e2 + e3);
  wts[t * 4 + 0] = e0 * inv; wts[t * 4 + 1] = e1 * inv;
  wts[t * 4 + 2] = e2 * inv; wts[t * 4 + 3] = e3 * inv;
}

// ---------------------------------------------------------------------------
// Main fused expert kernel. Block = (n-tile of 128, m, b); 256 threads = 8 waves.
// ---------------------------------------------------------------------------
__global__ __launch_bounds__(256)
void moe_main(const float* __restrict__ x,
              const __bf16* __restrict__ pW1, const __bf16* __restrict__ pW2,
              const __bf16* __restrict__ pW3,
              const float* __restrict__ b1, const float* __restrict__ b2,
              const float* __restrict__ b3,
              const float* __restrict__ wts, float* __restrict__ out) {
  extern __shared__ char smem[];
  const int tid = threadIdx.x, lane = tid & 31, w = tid >> 5;
  const int b = blockIdx.z, m = blockIdx.y, n0 = blockIdx.x * NTILE_;
  const int rofs = (lane >= 16) ? 8 : 0;

  // ---- stage X tile (C=512 x 128) as bf16 B-fragments in LDS --------------
  // B fragment tile (kt,nt): lane l<16 holds K kt*32+0..15 of column nt*16+l,
  // lane l>=16 holds K kt*32+16..31 of column nt*16+(l-16); bf16 pairs per dword.
  {
    size_t base = ((size_t)b * 512) * 4 * 4096 + (size_t)m * 4096 + n0;
    int nl = lane * 4, nt = nl >> 4, nn2 = nl & 15;
    for (int i = 0; i < 16; ++i) {
      int c0 = 4 * w + 32 * i;                  // channel group of 4
      v4f r0 = *(const v4f*)(x + base + (size_t)(c0 + 0) * 16384 + nl);
      v4f r1 = *(const v4f*)(x + base + (size_t)(c0 + 1) * 16384 + nl);
      v4f r2 = *(const v4f*)(x + base + (size_t)(c0 + 2) * 16384 + nl);
      v4f r3 = *(const v4f*)(x + base + (size_t)(c0 + 3) * 16384 + nl);
      #pragma unroll
      for (int p = 0; p < 2; ++p) {
        int k0 = c0 + 2 * p;
        int kk = k0 & 31;
        unsigned tileoff = (unsigned)(((k0 >> 5) * 8 + nt) << 10);
        unsigned v2 = (unsigned)((kk & 15) >> 1);
        unsigned lb = (unsigned)(nn2 + (kk & 16));
        #pragma unroll
        for (int j = 0; j < 4; ++j) {
          unsigned dw = p ? pack_bf16x2(r2[j], r3[j]) : pack_bf16x2(r0[j], r1[j]);
          *(unsigned*)(smem + XP_OFF + tileoff + (lb + j) * 32u + v2 * 4u) = dw;
        }
      }
    }
  }
  __syncthreads();

  float wsc[4];
  #pragma unroll
  for (int e = 0; e < 4; ++e) wsc[e] = wts[(b * 4 + m) * 4 + e];

  v8f oacc = {0.f, 0.f, 0.f, 0.f, 0.f, 0.f, 0.f, 0.f};

  for (int e = 0; e < E_; ++e) {
    // ---- GEMM1: h1(128x128) = W1(128x512) @ X; wave w owns rows w*16.. ----
    v8f acc[8];
    #pragma unroll
    for (int nt = 0; nt < 8; ++nt) acc[nt] = (v8f){0,0,0,0,0,0,0,0};
    const __bf16* a1p = pW1 + (size_t)((e * 8 + w) * 16) * 512 + lane * 16;
    for (int kt = 0; kt < 16; ++kt) {
      v16bf a = *(const v16bf*)(a1p + kt * 512);
      #pragma unroll
      for (int nt = 0; nt < 8; ++nt) {
        v16bf bb = *(const v16bf*)(smem + XP_OFF + (unsigned)(((kt * 8 + nt) << 10) + lane * 32));
        acc[nt] = wmma_bf16(a, bb, acc[nt]);
      }
    }
    // bias + relu + repack D->B fragments into H1P
    {
      unsigned kt2 = (unsigned)(w >> 1);
      unsigned l2  = (unsigned)((lane & 15) + ((w & 1) << 4));
      #pragma unroll
      for (int v = 0; v < 8; v += 2) {
        float bv0 = b1[e * 128 + w * 16 + v + rofs];
        float bv1 = b1[e * 128 + w * 16 + v + 1 + rofs];
        unsigned v2 = (unsigned)((v + rofs) >> 1);
        #pragma unroll
        for (int nt = 0; nt < 8; ++nt) {
          float f0 = fmaxf(acc[nt][v] + bv0, 0.f);
          float f1 = fmaxf(acc[nt][v + 1] + bv1, 0.f);
          *(unsigned*)(smem + H1P_OFF + (unsigned)(((kt2 * 8 + nt) << 10)) + l2 * 32u + v2 * 4u)
              = pack_bf16x2(f0, f1);
        }
      }
    }
    __syncthreads();

    // ---- GEMM2: h2(64x128, rows>=48 are zero-padded) = W2p @ h1 -----------
    const int mt  = w >> 1;
    const int ntb = (w & 1) * 4;
    v8f acc2[4];
    #pragma unroll
    for (int j = 0; j < 4; ++j) acc2[j] = (v8f){0,0,0,0,0,0,0,0};
    const __bf16* a2p = pW2 + (size_t)((e * 4 + mt) * 4) * 512 + lane * 16;
    for (int kt = 0; kt < 4; ++kt) {
      v16bf a = *(const v16bf*)(a2p + kt * 512);
      #pragma unroll
      for (int j = 0; j < 4; ++j) {
        v16bf bb = *(const v16bf*)(smem + H1P_OFF + (unsigned)(((kt * 8 + ntb + j) << 10) + lane * 32));
        acc2[j] = wmma_bf16(a, bb, acc2[j]);
      }
    }
    {
      unsigned kt2 = (unsigned)(mt >> 1);
      unsigned l2  = (unsigned)((lane & 15) + ((mt & 1) << 4));
      #pragma unroll
      for (int v = 0; v < 8; v += 2) {
        int row0 = mt * 16 + v + rofs;
        float bv0 = (row0 < 48) ? b2[e * 48 + row0] : 0.f;
        float bv1 = (row0 + 1 < 48) ? b2[e * 48 + row0 + 1] : 0.f;
        unsigned v2 = (unsigned)((v + rofs) >> 1);
        #pragma unroll
        for (int j = 0; j < 4; ++j) {
          float f0 = fmaxf(acc2[j][v] + bv0, 0.f);
          float f1 = fmaxf(acc2[j][v + 1] + bv1, 0.f);
          *(unsigned*)(smem + H2P_OFF + (unsigned)(((kt2 * 8 + ntb + j) << 10)) + l2 * 32u + v2 * 4u)
              = pack_bf16x2(f0, f1);
        }
      }
    }
    __syncthreads();

    // ---- GEMM3: y(16x128) = W3p(16x64) @ h2; wave w owns n-tile w ---------
    v8f acc3 = {0.f, 0.f, 0.f, 0.f, 0.f, 0.f, 0.f, 0.f};
    const __bf16* a3p = pW3 + (size_t)(e * 2) * 512 + lane * 16;
    #pragma unroll
    for (int kt = 0; kt < 2; ++kt) {
      v16bf a = *(const v16bf*)(a3p + kt * 512);
      v16bf bb = *(const v16bf*)(smem + H2P_OFF + (unsigned)(((kt * 8 + w) << 10) + lane * 32));
      acc3 = wmma_bf16(a, bb, acc3);
    }
    float we = wsc[e];
    #pragma unroll
    for (int v = 0; v < 8; ++v)
      oacc[v] += we * (acc3[v] + b3[e * 16 + v + rofs]);
  }

  // ---- store out[b, o, m, n] ------------------------------------------------
  size_t obase = (((size_t)b * 16) * 4 + m) * 4096 + n0 + w * 16 + (lane & 15);
  #pragma unroll
  for (int v = 0; v < 8; ++v)
    out[obase + (size_t)(v + rofs) * 16384] = oacc[v];
}

// ---------------------------------------------------------------------------
extern "C" void kernel_launch(void* const* d_in, const int* in_sizes, int n_in,
                              void* d_out, int out_size, void* d_ws, size_t ws_size,
                              hipStream_t stream) {
  const float* x   = (const float*)d_in[0];
  const float* W1  = (const float*)d_in[1];
  const float* b1  = (const float*)d_in[2];
  const float* W2  = (const float*)d_in[3];
  const float* b2  = (const float*)d_in[4];
  const float* W3  = (const float*)d_in[5];
  const float* b3  = (const float*)d_in[6];
  const float* Wg1 = (const float*)d_in[7];
  const float* bg1 = (const float*)d_in[8];
  const float* Wg2 = (const float*)d_in[9];
  const float* bg2 = (const float*)d_in[10];
  float* out = (float*)d_out;

  char* ws = (char*)d_ws;
  __bf16* pW1 = (__bf16*)(ws + WS_PW1);
  __bf16* pW2 = (__bf16*)(ws + WS_PW2);
  __bf16* pW3 = (__bf16*)(ws + WS_PW3);
  float* pooled = (float*)(ws + WS_POOL);
  float* gbuf   = (float*)(ws + WS_G);
  float* wts    = (float*)(ws + WS_WTS);

  // weight repack (bf16 A-fragments)
  pack_w<<<(E_ * 8 * 16 * 512 + 255) / 256, 256, 0, stream>>>(W1, pW1, 8, 16, H1_, C_);
  pack_w<<<(E_ * 4 * 4  * 512 + 255) / 256, 256, 0, stream>>>(W2, pW2, 4, 4,  H2_, H1_);
  pack_w<<<(E_ * 1 * 2  * 512 + 255) / 256, 256, 0, stream>>>(W3, pW3, 1, 2,  OUT_, H2_);

  // gate path
  pool_k<<<4096, 256, 0, stream>>>(x, pooled);
  gate_k<<<64, 128, 0, stream>>>(pooled, Wg1, bg1, Wg2, bg2, gbuf);
  softmax_k<<<1, 64, 0, stream>>>(gbuf, wts);

  // fused expert pipeline
  (void)hipFuncSetAttribute(reinterpret_cast<const void*>(moe_main),
                            hipFuncAttributeMaxDynamicSharedMemorySize,
                            (int)SMEM_BYTES);
  dim3 grid(N_ / NTILE_, M_, B_);
  moe_main<<<grid, 256, SMEM_BYTES, stream>>>(x, pW1, pW2, pW3, b1, b2, b3, wts, out);
}
